// DYN_GNN_69870527971993
// MI455X (gfx1250) — matrix operand order repaired and checked by reading, not compiled, not run on hardware
//
#include <hip/hip_runtime.h>
#include <hip/hip_bf16.h>
#include <math.h>

// Problem constants (match reference)
#define S_T     8
#define N_NODES 100000
#define E_EDGES 1600000
#define D_F     64
#define L_LAYERS 3
#define B_BATCH 128
#define HR_DIM  128
#define BN_EPSF 1e-5f

// f16 LDS tile stride: 64 data + 8 pad halves = 144B/row (4-dword bank
// rotation per row, keeps 16B alignment for b128 loads)
#define LDS_STRIDE 72
// f32 D-tile stride: 64 data + 4 pad floats = 272B/row (alignment 16B kept)
#define LDS_DSTR   68
// One prepped weight matrix in f16 elements / in 16B chunks
#define WMAT_ELEMS (64 * LDS_STRIDE)
#define WMAT_CHUNKS ((WMAT_ELEMS * 2) / 16)

#if defined(__gfx1250__) && __has_builtin(__builtin_amdgcn_global_load_async_to_lds_b128)
#define HAVE_ASYNC_LDS 1
#else
#define HAVE_ASYNC_LDS 0
#endif

typedef __attribute__((ext_vector_type(16))) _Float16 v16h;
typedef __attribute__((ext_vector_type(8)))  float    v8f;
typedef __attribute__((ext_vector_type(4)))  int      v4i_t;

union AFrag { v16h v; float4 f4[2]; _Float16 e[16]; };
union CFrag { v8f  v; float  e[8]; };
union H4U   { _Float16 h[4]; uint2 u2; };

// ---------------------------------------------------------------------------
// Edge gather/scatter: agg[dst] += x[src].  16 lanes per edge, float4 chunks.
// Dominant cost of the whole net: 24 passes x (~410MB gather + 102M f32
// atomics); agg (25.6MB) is resident in the 192MB L2 so atomics resolve there.
// ---------------------------------------------------------------------------
__global__ __launch_bounds__(256) void edge_scatter_kernel(
    const float* __restrict__ x, const int* __restrict__ src,
    const int* __restrict__ dst, float* __restrict__ agg)
{
    unsigned tid = blockIdx.x * 256u + threadIdx.x;
    unsigned e  = tid >> 4;
    unsigned ch = tid & 15u;
    if (e >= (unsigned)E_EDGES) return;
    int s = src[e], d = dst[e];
    const float4 v = reinterpret_cast<const float4*>(x + (size_t)s * 64)[ch];
    float* ap = agg + (size_t)d * 64 + ch * 4;
    atomicAdd(ap + 0, v.x);
    atomicAdd(ap + 1, v.y);
    atomicAdd(ap + 2, v.z);
    atomicAdd(ap + 3, v.w);
}

// ---------------------------------------------------------------------------
// One-shot weight prep: all 48 64x64 matrices -> f16, TRANSPOSED (col-major),
// padded to LDS_STRIDE, so GEMM kernels can async-DMA them straight into LDS.
// wprep[mat][col*LDS_STRIDE + k] = W[mat/2][k][col], mat = (t*L+l)*2 + {0,1}
// ---------------------------------------------------------------------------
__global__ __launch_bounds__(256) void weight_prep_kernel(
    const float* __restrict__ W1, const float* __restrict__ W2,
    _Float16* __restrict__ wprep)
{
    unsigned i = blockIdx.x * 256u + threadIdx.x;
    if (i >= 48u * 4096u) return;
    unsigned mat = i >> 12;
    unsigned rem = i & 4095u;
    unsigned k = rem >> 6, col = rem & 63u;
    unsigned tl = mat >> 1;
    const float* srcW = (mat & 1u) ? W2 : W1;
    wprep[(size_t)mat * WMAT_ELEMS + col * LDS_STRIDE + k] =
        (_Float16)srcW[(size_t)tl * 4096 + rem];
}

// Stage one prepped weight matrix into LDS (async DMA on gfx1250)
__device__ __forceinline__ void stage_weights(const _Float16* __restrict__ wsrc,
                                              _Float16* sWt, int tid)
{
    const uint4* gw = reinterpret_cast<const uint4*>(wsrc);
    uint4* lw = reinterpret_cast<uint4*>(sWt);
#if HAVE_ASYNC_LDS
    for (int i = tid; i < WMAT_CHUNKS; i += 256)
        __builtin_amdgcn_global_load_async_to_lds_b128(
            (__attribute__((address_space(1))) v4i_t*)(gw + i),
            (__attribute__((address_space(3))) v4i_t*)(lw + i), 0, 0);
#else
    for (int i = tid; i < WMAT_CHUNKS; i += 256) lw[i] = gw[i];
#endif
}

__device__ __forceinline__ void wait_async_stage()
{
#if HAVE_ASYNC_LDS
#if __has_builtin(__builtin_amdgcn_s_wait_asynccnt)
    __builtin_amdgcn_s_wait_asynccnt(0);
#else
    asm volatile("s_wait_asynccnt 0x0" ::: "memory");
#endif
#endif
}

// Build A fragment (16x32 f16, CDNA5 layout) from padded row-major LDS tile.
// Lane<16: K runs [base,base+8) and [base+16,base+24); lane>=16: +8.
__device__ __forceinline__ AFrag load_afrag(const _Float16* sA, int row,
                                            int ks, int lane)
{
    AFrag a;
    const int base = ks * 32 + ((lane < 16) ? 0 : 8);
    const _Float16* p = sA + row * LDS_STRIDE + base;
    a.f4[0] = *reinterpret_cast<const float4*>(p);
    a.f4[1] = *reinterpret_cast<const float4*>(p + 16);
    return a;
}

// Build B fragment (32x16 f16) from transposed LDS tile (sWt[col][k]):
// lane<16: K [kb,kb+8) and [kb+16,kb+24) for col=lane; lane>=16: +8 (matches A).
__device__ __forceinline__ AFrag load_bfrag(const _Float16* sWt, int col,
                                            int ks, int lane)
{
    AFrag b;
    const int kb = ks * 32 + ((lane < 16) ? 0 : 8);
    const _Float16* p = sWt + col * LDS_STRIDE + kb;
    b.f4[0] = *reinterpret_cast<const float4*>(p);
    b.f4[1] = *reinterpret_cast<const float4*>(p + 16);
    return b;
}

// ---------------------------------------------------------------------------
// GEMM1 fused: ha = (x + agg) @ W1 + b1, plus per-channel BN sum/sumsq.
// 256 threads = 8 waves; each wave -> 16 rows x 64 cols via 8 WMMA ops.
// Epilogue goes through an LDS f32 tile -> coalesced float4 stores; BN stats
// accumulate in registers (fixed column set per thread) -> 8 LDS atomics each.
// ---------------------------------------------------------------------------
__global__ __launch_bounds__(256) void gin_gemm1_kernel(
    const float* __restrict__ xin, const float* __restrict__ agg,
    const _Float16* __restrict__ wsrc, const float* __restrict__ b1,
    float* __restrict__ ha, float* __restrict__ stats)
{
    __shared__ __align__(16) _Float16 sA[128 * LDS_STRIDE];   // 18 KB
    __shared__ __align__(16) _Float16 sWt[64 * LDS_STRIDE];   //  9 KB
    __shared__ __align__(16) float    sD[128 * LDS_DSTR];     // 34 KB
    __shared__ float sSum[64], sSq[64];

    const int tid  = threadIdx.x;
    const int lane = tid & 31;
    const int wave = tid >> 5;
    const int rowBlock = blockIdx.x * 128;

    if (tid < 64) { sSum[tid] = 0.f; sSq[tid] = 0.f; }

    stage_weights(wsrc, sWt, tid);  // async DMA (ASYNCcnt) on gfx1250

    // Activation tile: float4 global reads -> packed 4xf16 8B LDS store
    for (int i = tid; i < 128 * 16; i += 256) {
        int r = i >> 4, c4 = (i & 15) * 4;
        int gr = rowBlock + r;
        H4U p;
        if (gr < N_NODES) {
            size_t o = (size_t)gr * 64 + c4;
            float4 xv = *reinterpret_cast<const float4*>(&xin[o]);
            float4 av = *reinterpret_cast<const float4*>(&agg[o]);
            p.h[0] = (_Float16)(xv.x + av.x);
            p.h[1] = (_Float16)(xv.y + av.y);
            p.h[2] = (_Float16)(xv.z + av.z);
            p.h[3] = (_Float16)(xv.w + av.w);
        } else {
            p.h[0] = p.h[1] = p.h[2] = p.h[3] = (_Float16)0.f;
        }
        *reinterpret_cast<uint2*>(&sA[r * LDS_STRIDE + c4]) = p.u2;
    }
    wait_async_stage();
    __syncthreads();

    const int rowBase = wave * 16;
    const int m = lane & 15;
    CFrag acc[4];
#pragma unroll
    for (int c = 0; c < 4; ++c)
#pragma unroll
        for (int j = 0; j < 8; ++j) acc[c].e[j] = 0.f;

#pragma unroll
    for (int ks = 0; ks < 2; ++ks) {          // K = 64 in two 32-chunks
        AFrag a = load_afrag(sA, rowBase + m, ks, lane);
#pragma unroll
        for (int c = 0; c < 4; ++c) {
            AFrag b = load_bfrag(sWt, c * 16 + m, ks, lane);
            acc[c].v = __builtin_amdgcn_wmma_f32_16x16x32_f16(
                false, a.v, false, b.v, (short)0, acc[c].v, false, false);
        }
    }

    // D layout: VGPR j -> M=j (lanes 0-15) / M=j+8 (lanes 16-31), N=lane&15
#pragma unroll
    for (int c = 0; c < 4; ++c)
#pragma unroll
        for (int j = 0; j < 8; ++j) {
            int mm = j + ((lane < 16) ? 0 : 8);
            sD[(rowBase + mm) * LDS_DSTR + c * 16 + m] = acc[c].e[j];
        }
    __syncthreads();

    // Coalesced writeout + register-accumulated BN stats
    {
        const int c4 = (tid & 15) * 4;   // invariant column set per thread
        float4 bv = *reinterpret_cast<const float4*>(&b1[c4]);
        float s0 = 0.f, s1 = 0.f, s2 = 0.f, s3 = 0.f;
        float q0 = 0.f, q1 = 0.f, q2 = 0.f, q3 = 0.f;
        for (int i = tid; i < 128 * 16; i += 256) {
            int r = i >> 4;
            int gr = rowBlock + r;
            if (gr < N_NODES) {
                const float* dp = &sD[r * LDS_DSTR + c4];
                float4 v;
                v.x = dp[0] + bv.x; v.y = dp[1] + bv.y;
                v.z = dp[2] + bv.z; v.w = dp[3] + bv.w;
                *reinterpret_cast<float4*>(&ha[(size_t)gr * 64 + c4]) = v;
                s0 += v.x; s1 += v.y; s2 += v.z; s3 += v.w;
                q0 += v.x * v.x; q1 += v.y * v.y;
                q2 += v.z * v.z; q3 += v.w * v.w;
            }
        }
        atomicAdd(&sSum[c4],     s0); atomicAdd(&sSum[c4 + 1], s1);
        atomicAdd(&sSum[c4 + 2], s2); atomicAdd(&sSum[c4 + 3], s3);
        atomicAdd(&sSq[c4],      q0); atomicAdd(&sSq[c4 + 1],  q1);
        atomicAdd(&sSq[c4 + 2],  q2); atomicAdd(&sSq[c4 + 3],  q3);
    }
    __syncthreads();
    if (tid < 64) {
        atomicAdd(&stats[tid],      sSum[tid]);
        atomicAdd(&stats[64 + tid], sSq[tid]);
    }
}

// stats[0..63]=sum, [64..127]=sumsq -> [128..191]=mu, [192..255]=rstd
__global__ void bn_finalize_kernel(float* __restrict__ stats)
{
    int c = threadIdx.x;
    if (c < 64) {
        float mu  = stats[c] / (float)N_NODES;
        float var = stats[64 + c] / (float)N_NODES - mu * mu;
        stats[128 + c] = mu;
        stats[192 + c] = rsqrtf(var + BN_EPSF);
    }
}

// ---------------------------------------------------------------------------
// GEMM2 fused: xout = tanh( tanh(BN(ha)) @ W2 + b2 )
// ---------------------------------------------------------------------------
__global__ __launch_bounds__(256) void gin_gemm2_kernel(
    const float* __restrict__ ha, const float* __restrict__ stats,
    const float* __restrict__ gamma, const float* __restrict__ beta,
    const _Float16* __restrict__ wsrc, const float* __restrict__ b2,
    float* __restrict__ xout)
{
    __shared__ __align__(16) _Float16 sA[128 * LDS_STRIDE];
    __shared__ __align__(16) _Float16 sWt[64 * LDS_STRIDE];
    __shared__ __align__(16) float    sD[128 * LDS_DSTR];

    const int tid  = threadIdx.x;
    const int lane = tid & 31;
    const int wave = tid >> 5;
    const int rowBlock = blockIdx.x * 128;

    stage_weights(wsrc, sWt, tid);

    for (int i = tid; i < 128 * 16; i += 256) {
        int r = i >> 4, c4 = (i & 15) * 4;
        int gr = rowBlock + r;
        H4U p;
        if (gr < N_NODES) {
            size_t o = (size_t)gr * 64 + c4;
            float4 hv = *reinterpret_cast<const float4*>(&ha[o]);
            float4 mu = *reinterpret_cast<const float4*>(&stats[128 + c4]);
            float4 rs = *reinterpret_cast<const float4*>(&stats[192 + c4]);
            float4 gm = *reinterpret_cast<const float4*>(&gamma[c4]);
            float4 bt = *reinterpret_cast<const float4*>(&beta[c4]);
            p.h[0] = (_Float16)tanhf((hv.x - mu.x) * rs.x * gm.x + bt.x);
            p.h[1] = (_Float16)tanhf((hv.y - mu.y) * rs.y * gm.y + bt.y);
            p.h[2] = (_Float16)tanhf((hv.z - mu.z) * rs.z * gm.z + bt.z);
            p.h[3] = (_Float16)tanhf((hv.w - mu.w) * rs.w * gm.w + bt.w);
        } else {
            p.h[0] = p.h[1] = p.h[2] = p.h[3] = (_Float16)0.f;
        }
        *reinterpret_cast<uint2*>(&sA[r * LDS_STRIDE + c4]) = p.u2;
    }
    wait_async_stage();
    __syncthreads();

    const int rowBase = wave * 16;
    const int m = lane & 15;
    CFrag acc[4];
#pragma unroll
    for (int c = 0; c < 4; ++c)
#pragma unroll
        for (int j = 0; j < 8; ++j) acc[c].e[j] = 0.f;

#pragma unroll
    for (int ks = 0; ks < 2; ++ks) {
        AFrag a = load_afrag(sA, rowBase + m, ks, lane);
#pragma unroll
        for (int c = 0; c < 4; ++c) {
            AFrag b = load_bfrag(sWt, c * 16 + m, ks, lane);
            acc[c].v = __builtin_amdgcn_wmma_f32_16x16x32_f16(
                false, a.v, false, b.v, (short)0, acc[c].v, false, false);
        }
    }

#pragma unroll
    for (int c = 0; c < 4; ++c)
#pragma unroll
        for (int j = 0; j < 8; ++j) {
            int mm = j + ((lane < 16) ? 0 : 8);
            sD[(rowBase + mm) * LDS_DSTR + c * 16 + m] = acc[c].e[j];
        }
    __syncthreads();

    {
        const int c4 = (tid & 15) * 4;
        float4 bv = *reinterpret_cast<const float4*>(&b2[c4]);
        for (int i = tid; i < 128 * 16; i += 256) {
            int r = i >> 4;
            int gr = rowBlock + r;
            if (gr < N_NODES) {
                const float* dp = &sD[r * LDS_DSTR + c4];
                float4 v;
                v.x = tanhf(dp[0] + bv.x); v.y = tanhf(dp[1] + bv.y);
                v.z = tanhf(dp[2] + bv.z); v.w = tanhf(dp[3] + bv.w);
                *reinterpret_cast<float4*>(&xout[(size_t)gr * 64 + c4]) = v;
            }
        }
    }
}

// global_add_pool: g[batch[n]] += x[n]
__global__ __launch_bounds__(256) void pool_kernel(
    const float* __restrict__ x, const int* __restrict__ batch,
    float* __restrict__ g)
{
    unsigned tid = blockIdx.x * 256u + threadIdx.x;
    if (tid >= (unsigned)N_NODES * 64u) return;
    unsigned node = tid >> 6, c = tid & 63u;
    int seg = batch[node];
    atomicAdd(&g[(size_t)seg * 64 + c], x[(size_t)node * 64 + c]);
}

// hseq[t,b,:] = g[t,b,:] @ Wg + bg
__global__ __launch_bounds__(256) void proj_kernel(
    const float* __restrict__ g, const float* __restrict__ Wg,
    const float* __restrict__ bg, float* __restrict__ hseq)
{
    unsigned tid = blockIdx.x * 256u + threadIdx.x;
    if (tid >= (unsigned)(S_T * B_BATCH * 64)) return;
    unsigned o  = tid & 63u;
    unsigned tb = tid >> 6;
    const float* gr = g + (size_t)tb * 64;
    float s = bg[o];
    for (int k = 0; k < 64; ++k) s += gr[k] * Wg[k * 64 + o];
    hseq[tid] = s;
}

// Vanilla RNN over S=8 steps + sigmoid readout; single block, h in LDS.
__global__ __launch_bounds__(128) void rnn_kernel(
    const float* __restrict__ hseq, const float* __restrict__ Wih,
    const float* __restrict__ Whh,  const float* __restrict__ bih,
    const float* __restrict__ bhh,  const float* __restrict__ Wr,
    const float* __restrict__ br,   float* __restrict__ out)
{
    __shared__ float sh[B_BATCH * HR_DIM];  // 64 KB (320 KB LDS available)
    int b = threadIdx.x;
    for (int k = 0; k < HR_DIM; ++k) sh[b * HR_DIM + k] = 0.f;
    __syncthreads();

    float hn[HR_DIM];
    for (int t = 0; t < S_T; ++t) {
        const float* xt = hseq + ((size_t)t * B_BATCH + b) * 64;
        const float* hb = sh + b * HR_DIM;
        for (int j = 0; j < HR_DIM; ++j) {
            float s = bih[j] + bhh[j];
            const float* wi = Wih + (size_t)j * 64;
            for (int k = 0; k < 64; ++k) s += xt[k] * wi[k];
            const float* wh = Whh + (size_t)j * HR_DIM;
            for (int k = 0; k < HR_DIM; ++k) s += hb[k] * wh[k];
            hn[j] = tanhf(s);
        }
        __syncthreads();
        for (int j = 0; j < HR_DIM; ++j) sh[b * HR_DIM + j] = hn[j];
        __syncthreads();
    }
    float s = br[0];
    for (int k = 0; k < HR_DIM; ++k) s += sh[b * HR_DIM + k] * Wr[k];
    out[b] = 1.f / (1.f + expf(-s));
}

// ---------------------------------------------------------------------------
extern "C" void kernel_launch(void* const* d_in, const int* in_sizes, int n_in,
                              void* d_out, int out_size, void* d_ws, size_t ws_size,
                              hipStream_t stream)
{
    (void)in_sizes; (void)n_in; (void)out_size; (void)ws_size;
    const float* xs    = (const float*)d_in[0];
    const int*   ei    = (const int*)  d_in[1];
    const int*   batch = (const int*)  d_in[2];
    const float* W1    = (const float*)d_in[3];
    const float* b1    = (const float*)d_in[4];
    const float* gamma = (const float*)d_in[5];
    const float* beta  = (const float*)d_in[6];
    const float* W2    = (const float*)d_in[7];
    const float* b2    = (const float*)d_in[8];
    const float* Wg    = (const float*)d_in[9];
    const float* bg    = (const float*)d_in[10];
    const float* Wih   = (const float*)d_in[11];
    const float* Whh   = (const float*)d_in[12];
    const float* bih   = (const float*)d_in[13];
    const float* bhh   = (const float*)d_in[14];
    const float* Wr    = (const float*)d_in[15];
    const float* br    = (const float*)d_in[16];
    float* out = (float*)d_out;

    // Workspace layout (floats): 4 x N*64 tensors + stats + pooled/hseq + f16
    // prepped weights ≈ 103 MB
    float* ws   = (float*)d_ws;
    const size_t NF = (size_t)N_NODES * 64;
    float* bufA  = ws;
    float* bufB  = ws + NF;
    float* agg   = ws + 2 * NF;
    float* ha    = ws + 3 * NF;
    float* stats = ws + 4 * NF;       // 256 floats
    float* g     = stats + 256;       // S*B*64
    float* hseq  = g + (size_t)S_T * B_BATCH * 64;
    _Float16* wprep = (_Float16*)(hseq + (size_t)S_T * B_BATCH * 64);

    (void)hipMemsetAsync(g, 0, (size_t)S_T * B_BATCH * 64 * sizeof(float), stream);

    dim3 blk(256);
    const int gemmGrid = (N_NODES + 127) / 128;
    const int scatGrid = (E_EDGES * 16 + 255) / 256;
    const int poolGrid = (N_NODES * 64 + 255) / 256;

    weight_prep_kernel<<<(48 * 4096 + 255) / 256, blk, 0, stream>>>(W1, W2, wprep);

    for (int t = 0; t < S_T; ++t) {
        for (int l = 0; l < L_LAYERS; ++l) {
            const float* lin  = (l == 0) ? (xs + (size_t)t * NF)
                              : (l == 1) ? bufA : bufB;
            float*       lout = (l == 1) ? bufB : bufA;
            const int* src = ei + (size_t)t * 2 * E_EDGES;
            const int* dst = src + E_EDGES;
            const size_t tl = (size_t)t * L_LAYERS + l;

            (void)hipMemsetAsync(agg,   0, NF * sizeof(float),  stream);
            (void)hipMemsetAsync(stats, 0, 128 * sizeof(float), stream);
            edge_scatter_kernel<<<scatGrid, blk, 0, stream>>>(lin, src, dst, agg);
            gin_gemm1_kernel<<<gemmGrid, blk, 0, stream>>>(
                lin, agg, wprep + (tl * 2 + 0) * WMAT_ELEMS, b1 + tl * 64,
                ha, stats);
            bn_finalize_kernel<<<1, 64, 0, stream>>>(stats);
            gin_gemm2_kernel<<<gemmGrid, blk, 0, stream>>>(
                ha, stats, gamma + tl * 64, beta + tl * 64,
                wprep + (tl * 2 + 1) * WMAT_ELEMS, b2 + tl * 64, lout);
        }
        pool_kernel<<<poolGrid, blk, 0, stream>>>(
            bufA, batch, g + (size_t)t * B_BATCH * 64);
    }

    proj_kernel<<<(S_T * B_BATCH * 64 + 255) / 256, blk, 0, stream>>>(g, Wg, bg, hseq);
    rnn_kernel<<<1, 128, 0, stream>>>(hseq, Wih, Whh, bih, bhh, Wr, br, out);
}
